// RSSMCell_61529701482720
// MI455X (gfx1250) — matrix operand order repaired and checked by reading, not compiled, not run on hardware
//
#include <hip/hip_runtime.h>

// ---------------------------------------------------------------------------
// RSSM cell forward for gfx1250 (MI455X):
//   bf16 WMMA GEMMs, TDM (tensor_load_to_lds) double-buffered LDS staging,
//   fused LayerNorm-GRU + distribution-head pointwise kernels.
// ---------------------------------------------------------------------------

typedef __attribute__((ext_vector_type(16))) __bf16 v16bf;
typedef __attribute__((ext_vector_type(8)))  float  v8f;
typedef __attribute__((ext_vector_type(4)))  unsigned int u32x4;
typedef __attribute__((ext_vector_type(8)))  int    i32x8;
typedef __attribute__((ext_vector_type(4)))  int    i32x4;
typedef unsigned short bf16_t;
typedef unsigned int   u32;

#define BATCH 16384
#define EMBED 1536
#define DETER 1024
#define HID   1024

static __device__ __forceinline__ bf16_t f2bf(float f) {
  u32 u = __builtin_bit_cast(u32, f);
  u32 r = u + 0x7FFFu + ((u >> 16) & 1u);   // round-to-nearest-even
  return (bf16_t)(r >> 16);
}
static __device__ __forceinline__ float sigmoidf_(float x) {
  return 1.0f / (1.0f + __expf(-x));
}
static __device__ __forceinline__ float softplusf_(float x) {
  return (x > 15.0f) ? x : log1pf(__expf(x));
}

// ---------------------------------------------------------------------------
// TDM: async 2D tile DMA global -> LDS (bf16 elements).
// Tile: tile_rows x 64 elems; LDS rows padded 128B + 16B (LSTR=72 elems) via
// D# pad fields (pad_interval=4 -> 32 DWORDs, pad_amount=3 -> 4 DWORDs).
// rem_k / rem_rows are the remaining tensor extent measured from the tile
// origin; TDM zero-fills out-of-bounds reads (used by the N=64 head GEMMs).
// This toolchain exposes the 6-arg builtin:
//   (u32x4 g0, i32x8 g1, i32x4 g2, i32x4 g3, i32x8 g4, i32 cpol)
// ---------------------------------------------------------------------------
static __device__ __forceinline__ void tdm_load_2d(
    unsigned int lds_off, const bf16_t* gptr,
    unsigned int rem_k, unsigned int rem_rows, unsigned int tile_rows,
    unsigned long long row_stride_elems) {
  unsigned long long ga = (unsigned long long)(size_t)gptr;

  u32x4 g0;
  g0.x = 1u;                                                    // count=1 (valid D#)
  g0.y = lds_off;                                               // lds_addr (bytes)
  g0.z = (unsigned int)ga;                                      // global_addr[31:0]
  g0.w = (unsigned int)((ga >> 32) & 0x01FFFFFFu) | 0x80000000u; // addr[56:32] | type=2

  i32x8 g1;
  g1[0] = (int)((1u << 16)      // data_size = 2 bytes
              | (1u << 20)      // pad_enable
              | (4u << 22)      // pad_interval: 32 DWORDs (one 128B row)
              | (3u << 25));    // pad_amount: 4 DWORDs (16B) -> LSTR=72 elems
  g1[1] = (int)((rem_k & 0xFFFFu) << 16);                            // tensor_dim0[15:0]
  g1[2] = (int)(((rem_k >> 16) & 0xFFFFu) | ((rem_rows & 0xFFFFu) << 16)); // dim0 hi | dim1 lo
  g1[3] = (int)(((rem_rows >> 16) & 0xFFFFu) | (64u << 16));         // dim1 hi | tile_dim0=64
  g1[4] = (int)(tile_rows & 0xFFFFu);                                // tile_dim1 | tile_dim2=0
  g1[5] = (int)(unsigned int)(row_stride_elems & 0xFFFFFFFFull);     // dim0_stride[31:0]
  g1[6] = (int)(unsigned int)((row_stride_elems >> 32) & 0xFFFFull); // dim0_stride[47:32]
  g1[7] = 0;                                                         // dim1_stride (unused, 2D)

  i32x4 z4 = {0, 0, 0, 0};
  i32x8 z8 = {0, 0, 0, 0, 0, 0, 0, 0};
  __builtin_amdgcn_tensor_load_to_lds(g0, g1, z4, z4, z8, 0);
}

// ---------------------------------------------------------------------------
// Prep / conversion kernels
// ---------------------------------------------------------------------------

__global__ void k_cvt_bf16(const float* __restrict__ in, bf16_t* __restrict__ out, int n) {
  int i = blockIdx.x * blockDim.x + threadIdx.x;
  if (i < n) out[i] = f2bf(in[i]);
}

// W_ru[n, k] : n in [0,2048) covers r then u gates; k<1024 -> w_ih[n,k], else w_hh[n,k-1024]
__global__ void k_pack_ru(const float* __restrict__ w_ih, const float* __restrict__ w_hh,
                          bf16_t* __restrict__ out) {
  int i = blockIdx.x * blockDim.x + threadIdx.x;
  if (i >= 2048 * 2048) return;
  int n = i >> 11, k = i & 2047;
  float v = (k < 1024) ? w_ih[n * 1024 + k] : w_hh[n * 1024 + (k - 1024)];
  out[i] = f2bf(v);
}

// w_pre [1024,44] -> padded [1024,64] bf16
__global__ void k_pack_wpre(const float* __restrict__ w_pre, bf16_t* __restrict__ out) {
  int i = blockIdx.x * blockDim.x + threadIdx.x;
  if (i >= 1024 * 64) return;
  int n = i >> 6, k = i & 63;
  float v = (k < 44) ? w_pre[n * 44 + k] : 0.0f;
  out[i] = f2bf(v);
}

// xin[b, 0:32]=stoc*mask, [32:44]=action, [44:64]=0  (bf16)
__global__ void k_prep_x(const float* __restrict__ stoc, const float* __restrict__ action,
                         const float* __restrict__ mask, bf16_t* __restrict__ out) {
  int i = blockIdx.x * blockDim.x + threadIdx.x;
  if (i >= BATCH * 64) return;
  int row = i >> 6, c = i & 63;
  float v = 0.0f;
  if (c < 32)      v = stoc[row * 32 + c] * mask[row];
  else if (c < 44) v = action[row * 12 + (c - 32)];
  out[i] = f2bf(v);
}

// deter*mask -> fp32 copy + bf16 copy
__global__ void k_prep_deter(const float* __restrict__ deter, const float* __restrict__ mask,
                             float* __restrict__ dm, bf16_t* __restrict__ db) {
  int i = blockIdx.x * blockDim.x + threadIdx.x;
  if (i >= BATCH * DETER) return;
  float v = deter[i] * mask[i >> 10];
  dm[i] = v;
  db[i] = f2bf(v);
}

// ---------------------------------------------------------------------------
// Generic bf16 GEMM:  C[M,N] = act( A[M,Ktot] @ W[N,Ktot]^T + bias )
// A given as split-K pair (A0:K0 cols | A1:K1 cols); segments are BK-aligned.
// EPI==0: store fp32.  EPI==1: ReLU then store bf16.
// Tile: BM=128, BN=128, BK=64; 8 waves, each wave 32x64 (2x4 WMMA frags).
// Staging: TDM double-buffered; each wave DMAs its 16-row slice of next
// A/W tiles while computing current stage; s_wait_tensorcnt + 1 barrier/iter.
// ---------------------------------------------------------------------------

#define BM 128
#define BN 128
#define BK 64
#define LSTR 72                   // LDS row stride (elems) = 128B row + 16B pad
#define STAGE_ELEMS (BM * LSTR)   // one tile (A or W) = 9216 elems = 18432 B
#define SMEM_BYTES (4 * STAGE_ELEMS * 2)  // 2 stages x (A + W) = 73728 B

template <int EPI>
__global__ __launch_bounds__(256)
void k_gemm_bf16(const bf16_t* __restrict__ A0, int K0,
                 const bf16_t* __restrict__ A1, int K1,
                 const bf16_t* __restrict__ W, const float* __restrict__ bias,
                 void* __restrict__ Cp, int M, int N) {
  extern __shared__ bf16_t smem[];
  // Stage layout (computed arithmetically; no pointer aggregates from LDS
  // symbols -- those become unsupported addrspacecast static initializers):
  //   stage s: A tile at smem + s*2*STAGE_ELEMS, W tile at +STAGE_ELEMS more.

  const int tid  = threadIdx.x;
  const int m0   = blockIdx.y * BM;
  const int n0   = blockIdx.x * BN;
  const int Ktot = K0 + K1;
  const int nK   = Ktot / BK;
  const int lane = tid & 31;
  const int wid  = tid >> 5;
  const int wm   = (wid & 3) * 32;   // wave M offset within tile
  const int wn   = (wid >> 2) * 64;  // wave N offset within tile
  const int lh   = lane >> 4;        // half-wave (0/1)
  const int l15  = lane & 15;
  const int slice = wid * 16;        // this wave's 16-row DMA slice

  // Issue the two 16-row tile-slice DMAs (A + W) for K-chunk kb into stage s.
  auto issue = [&](int kb, int s) {
    bf16_t* As = smem + s * (2 * STAGE_ELEMS);
    bf16_t* Ws = As + STAGE_ELEMS;
    const bf16_t* Aseg; int ksegoff, Kseg;
    if (kb < K0) { Aseg = A0; ksegoff = kb;      Kseg = K0; }
    else         { Aseg = A1; ksegoff = kb - K0; Kseg = K1; }
    tdm_load_2d((unsigned int)(size_t)(As + slice * LSTR),
                Aseg + (size_t)(m0 + slice) * Kseg + ksegoff,
                (unsigned int)(Kseg - ksegoff),
                (unsigned int)(M - (m0 + slice)),
                16u, (unsigned long long)Kseg);
    int wrow = n0 + slice;
    int remw = N - wrow; if (remw < 0) remw = 0;   // OOB rows -> TDM zero-fill
    tdm_load_2d((unsigned int)(size_t)(Ws + slice * LSTR),
                W + (size_t)wrow * Ktot + kb,
                (unsigned int)(Ktot - kb),
                (unsigned int)remw,
                16u, (unsigned long long)Ktot);
  };

  v8f acc[2][4];
#pragma unroll
  for (int i = 0; i < 2; ++i)
#pragma unroll
    for (int j = 0; j < 4; ++j)
      acc[i][j] = v8f{0.f, 0.f, 0.f, 0.f, 0.f, 0.f, 0.f, 0.f};

  // Prologue: stage 0 DMA, wait, publish.
  issue(0, 0);
  __builtin_amdgcn_s_wait_tensorcnt(0);
  __syncthreads();

  for (int t = 0; t < nK; ++t) {
    const int s = t & 1;
    if (t + 1 < nK) issue((t + 1) * BK, s ^ 1);   // prefetch next stage via TDM

    const bf16_t* Asm = smem + s * (2 * STAGE_ELEMS);
    const bf16_t* Wsm = Asm + STAGE_ELEMS;
#pragma unroll
    for (int ks = 0; ks < BK; ks += 32) {
      union { uint4 q[2]; v16bf v; } af[2], bfr[4];
      // A frag 16x32: lanes0-15 K{0..7,16..23}, lanes16-31 K{8..15,24..31}
#pragma unroll
      for (int i = 0; i < 2; ++i) {
        const bf16_t* p = &Asm[(wm + i * 16 + l15) * LSTR + ks + lh * 8];
        af[i].q[0] = *(const uint4*)(p);
        af[i].q[1] = *(const uint4*)(p + 16);
      }
      // B frag 32x16: lane = column; lanes0-15 K{0..7,16..23}, lanes16-31 K{8..15,24..31}
#pragma unroll
      for (int j = 0; j < 4; ++j) {
        const bf16_t* p = &Wsm[(wn + j * 16 + l15) * LSTR + ks + lh * 8];
        bfr[j].q[0] = *(const uint4*)(p);
        bfr[j].q[1] = *(const uint4*)(p + 16);
      }
#pragma unroll
      for (int i = 0; i < 2; ++i)
#pragma unroll
        for (int j = 0; j < 4; ++j)
          acc[i][j] = __builtin_amdgcn_wmma_f32_16x16x32_bf16(
              false, af[i].v, false, bfr[j].v, (short)0, acc[i][j], false, false);
    }
    // Wave's next-stage DMA complete + all waves done reading stage s.
    __builtin_amdgcn_s_wait_tensorcnt(0);
    __syncthreads();
  }

  // ---- epilogue: C/D layout: lane=column, VGPR r -> row (m + lh*8 + r) ----
#pragma unroll
  for (int i = 0; i < 2; ++i) {
#pragma unroll
    for (int j = 0; j < 4; ++j) {
      int nc = n0 + wn + j * 16 + l15;
      if (nc >= N) continue;
      float bv = bias ? bias[nc] : 0.0f;
#pragma unroll
      for (int r = 0; r < 8; ++r) {
        int   mr = m0 + wm + i * 16 + lh * 8 + r;
        float v  = acc[i][j][r] + bv;
        if (EPI == 1) {
          v = v > 0.0f ? v : 0.0f;
          ((bf16_t*)Cp)[(size_t)mr * N + nc] = f2bf(v);
        } else {
          ((float*)Cp)[(size_t)mr * N + nc] = v;
        }
      }
    }
  }
}

// ---------------------------------------------------------------------------
// LayerNorm-GRU pointwise: one 256-thread block per batch row (1024 elems).
// ---------------------------------------------------------------------------

__global__ __launch_bounds__(256)
void k_gru(const float* __restrict__ ru, const float* __restrict__ ni,
           const float* __restrict__ nh, const float* __restrict__ det,
           const float* __restrict__ lrg, const float* __restrict__ lrb,
           const float* __restrict__ lug, const float* __restrict__ lub,
           const float* __restrict__ lng, const float* __restrict__ lnb,
           float* __restrict__ h_out, bf16_t* __restrict__ h_b) {
  __shared__ float red[8][4];
  const int row = blockIdx.x;
  const int t = threadIdx.x, lane = t & 31, wid = t >> 5;

  float r[4], u[4], niv[4], nhv[4], dv[4];
  float sr = 0.f, sr2 = 0.f, su = 0.f, su2 = 0.f;
#pragma unroll
  for (int i = 0; i < 4; ++i) {
    int j  = t + i * 256;
    r[i]   = ru[(size_t)row * 2048 + j];
    u[i]   = ru[(size_t)row * 2048 + 1024 + j];
    niv[i] = ni[(size_t)row * 1024 + j];
    nhv[i] = nh[(size_t)row * 1024 + j];
    dv[i]  = det[(size_t)row * 1024 + j];
    sr += r[i]; sr2 += r[i] * r[i];
    su += u[i]; su2 += u[i] * u[i];
  }
#pragma unroll
  for (int m = 16; m > 0; m >>= 1) {
    sr += __shfl_xor(sr, m, 32);  sr2 += __shfl_xor(sr2, m, 32);
    su += __shfl_xor(su, m, 32);  su2 += __shfl_xor(su2, m, 32);
  }
  if (lane == 0) { red[wid][0] = sr; red[wid][1] = sr2; red[wid][2] = su; red[wid][3] = su2; }
  __syncthreads();
  sr = sr2 = su = su2 = 0.f;
#pragma unroll
  for (int w = 0; w < 8; ++w) { sr += red[w][0]; sr2 += red[w][1]; su += red[w][2]; su2 += red[w][3]; }

  const float inv = 1.0f / 1024.0f;
  float mr = sr * inv, vr = sr2 * inv - mr * mr;
  float mu = su * inv, vu = su2 * inv - mu * mu;
  float rsr = rsqrtf(vr + 1e-3f), rsu = rsqrtf(vu + 1e-3f);

  float tn[4], upd[4], st = 0.f, st2 = 0.f;
#pragma unroll
  for (int i = 0; i < 4; ++i) {
    int j = t + i * 256;
    float reset = sigmoidf_((r[i] - mr) * rsr * lrg[j] + lrb[j]);
    upd[i]      = sigmoidf_((u[i] - mu) * rsu * lug[j] + lub[j]);
    tn[i]       = niv[i] + reset * nhv[i];
    st += tn[i]; st2 += tn[i] * tn[i];
  }
#pragma unroll
  for (int m = 16; m > 0; m >>= 1) { st += __shfl_xor(st, m, 32); st2 += __shfl_xor(st2, m, 32); }
  __syncthreads();
  if (lane == 0) { red[wid][0] = st; red[wid][1] = st2; }
  __syncthreads();
  st = st2 = 0.f;
#pragma unroll
  for (int w = 0; w < 8; ++w) { st += red[w][0]; st2 += red[w][1]; }
  float mn = st * inv, vn = st2 * inv - mn * mn, rsn = rsqrtf(vn + 1e-3f);

#pragma unroll
  for (int i = 0; i < 4; ++i) {
    int j = t + i * 256;
    float newv = tanhf((tn[i] - mn) * rsn * lng[j] + lnb[j]);
    float h    = upd[i] * newv + (1.0f - upd[i]) * dv[i];
    h_out[(size_t)row * 1024 + j] = h;
    h_b[(size_t)row * 1024 + j]   = f2bf(h);
  }
}

// ---------------------------------------------------------------------------
// Heads epilogue: split mean/std, softplus+0.1, write 5 small outputs.
// ---------------------------------------------------------------------------

__global__ void k_heads(const float* __restrict__ post, const float* __restrict__ prior,
                        float* __restrict__ out) {
  int i = blockIdx.x * blockDim.x + threadIdx.x;
  if (i >= BATCH * 32) return;
  int row = i >> 5, c = i & 31;
  const size_t S = (size_t)BATCH * 32;
  float pm = post[(size_t)row * 64 + c];
  float ps = post[(size_t)row * 64 + 32 + c];
  float rm = prior[(size_t)row * 64 + c];
  float rs = prior[(size_t)row * 64 + 32 + c];
  out[i]                                = pm;                     // post_mean
  out[S + i]                            = softplusf_(ps) + 0.1f;  // post_std
  out[2 * S + i]                        = rm;                     // prior_mean
  out[3 * S + i]                        = softplusf_(rs) + 0.1f;  // prior_std
  out[4 * S + (size_t)BATCH * 1024 + i] = pm;                     // post_latent
}

// ---------------------------------------------------------------------------
// Launch
// ---------------------------------------------------------------------------

extern "C" void kernel_launch(void* const* d_in, const int* in_sizes, int n_in,
                              void* d_out, int out_size, void* d_ws, size_t ws_size,
                              hipStream_t stream) {
  const float* embedding = (const float*)d_in[0];
  const float* action    = (const float*)d_in[1];
  const float* mask      = (const float*)d_in[2];
  const float* deter     = (const float*)d_in[3];
  const float* stoc      = (const float*)d_in[4];
  const float* w_pre     = (const float*)d_in[5];
  const float* b_pre     = (const float*)d_in[6];
  const float* w_ih      = (const float*)d_in[7];
  const float* w_hh      = (const float*)d_in[8];
  const float* ln_r_g    = (const float*)d_in[9];
  const float* ln_r_b    = (const float*)d_in[10];
  const float* ln_u_g    = (const float*)d_in[11];
  const float* ln_u_b    = (const float*)d_in[12];
  const float* ln_n_g    = (const float*)d_in[13];
  const float* ln_n_b    = (const float*)d_in[14];
  const float* wp1       = (const float*)d_in[15];
  const float* bp1       = (const float*)d_in[16];
  const float* wp2       = (const float*)d_in[17];
  const float* bp2       = (const float*)d_in[18];
  const float* wq1       = (const float*)d_in[19];
  const float* bq1       = (const float*)d_in[20];
  const float* wq2       = (const float*)d_in[21];
  const float* bq2       = (const float*)d_in[22];
  float* out = (float*)d_out;

  // bump allocator over d_ws
  size_t off = 0;
  auto alloc = [&](size_t nbytes) -> void* {
    void* p = (char*)d_ws + off;
    off += (nbytes + 255) & ~(size_t)255;
    return p;
  };
  const size_t B = BATCH;

  bf16_t* emb_b   = (bf16_t*)alloc(B * EMBED * 2);
  bf16_t* wp1_b   = (bf16_t*)alloc((size_t)HID * DETER * 2);
  bf16_t* wq1_b   = (bf16_t*)alloc((size_t)HID * (DETER + EMBED) * 2);
  bf16_t* wp2_b   = (bf16_t*)alloc((size_t)64 * HID * 2);
  bf16_t* wq2_b   = (bf16_t*)alloc((size_t)64 * HID * 2);
  bf16_t* wihn_b  = (bf16_t*)alloc((size_t)DETER * HID * 2);
  bf16_t* whhn_b  = (bf16_t*)alloc((size_t)DETER * DETER * 2);
  bf16_t* wru_b   = (bf16_t*)alloc((size_t)2048 * 2048 * 2);
  bf16_t* wpre_b  = (bf16_t*)alloc((size_t)HID * 64 * 2);
  bf16_t* xin_b   = (bf16_t*)alloc(B * 64 * 2);
  float*  deter_m = (float*) alloc(B * DETER * 4);
  bf16_t* deter_b = (bf16_t*)alloc(B * DETER * 2);
  bf16_t* x_b     = (bf16_t*)alloc(B * HID * 2);
  float*  ru      = (float*) alloc(B * 2048 * 4);
  float*  ni      = (float*) alloc(B * DETER * 4);
  float*  nh      = (float*) alloc(B * DETER * 4);
  bf16_t* h_b     = (bf16_t*)alloc(B * DETER * 2);
  bf16_t* p1_b    = (bf16_t*)alloc(B * HID * 2);
  bf16_t* q1_b    = (bf16_t*)alloc(B * HID * 2);
  float*  prior   = (float*) alloc(B * 64 * 4);
  float*  post    = (float*) alloc(B * 64 * 4);

  auto g1 = [](size_t n) { return dim3((unsigned)((n + 255) / 256)); };

  // --- weight / input conversions ---
  k_cvt_bf16<<<g1(B * EMBED), 256, 0, stream>>>(embedding, emb_b, (int)(B * EMBED));
  k_cvt_bf16<<<g1(HID * DETER), 256, 0, stream>>>(wp1, wp1_b, HID * DETER);
  k_cvt_bf16<<<g1((size_t)HID * (DETER + EMBED)), 256, 0, stream>>>(wq1, wq1_b, HID * (DETER + EMBED));
  k_cvt_bf16<<<g1(64 * HID), 256, 0, stream>>>(wp2, wp2_b, 64 * HID);
  k_cvt_bf16<<<g1(64 * HID), 256, 0, stream>>>(wq2, wq2_b, 64 * HID);
  k_cvt_bf16<<<g1(DETER * HID), 256, 0, stream>>>(w_ih + (size_t)2048 * 1024, wihn_b, DETER * HID);
  k_cvt_bf16<<<g1(DETER * DETER), 256, 0, stream>>>(w_hh + (size_t)2048 * 1024, whhn_b, DETER * DETER);
  k_pack_ru<<<g1((size_t)2048 * 2048), 256, 0, stream>>>(w_ih, w_hh, wru_b);
  k_pack_wpre<<<g1(HID * 64), 256, 0, stream>>>(w_pre, wpre_b);
  k_prep_x<<<g1(B * 64), 256, 0, stream>>>(stoc, action, mask, xin_b);
  k_prep_deter<<<g1(B * DETER), 256, 0, stream>>>(deter, mask, deter_m, deter_b);

  // --- x = relu(pre(xin)) : [B,64] x [1024,64]^T -> bf16 [B,1024] ---
  k_gemm_bf16<1><<<dim3(HID / BN, BATCH / BM), 256, SMEM_BYTES, stream>>>(
      xin_b, 64, nullptr, 0, wpre_b, b_pre, x_b, BATCH, HID);

  // --- rsum|usum = [x|deter] @ W_ru^T : K=2048, N=2048 -> fp32 [B,2048] ---
  k_gemm_bf16<0><<<dim3(2048 / BN, BATCH / BM), 256, SMEM_BYTES, stream>>>(
      x_b, HID, deter_b, DETER, wru_b, nullptr, ru, BATCH, 2048);

  // --- ni = x @ w_ih_n^T ; nh = deter @ w_hh_n^T ---
  k_gemm_bf16<0><<<dim3(DETER / BN, BATCH / BM), 256, SMEM_BYTES, stream>>>(
      x_b, HID, nullptr, 0, wihn_b, nullptr, ni, BATCH, DETER);
  k_gemm_bf16<0><<<dim3(DETER / BN, BATCH / BM), 256, SMEM_BYTES, stream>>>(
      deter_b, DETER, nullptr, 0, whhn_b, nullptr, nh, BATCH, DETER);

  // --- GRU pointwise: h (fp32 into d_out) + h_bf16 ---
  float* h_out = out + (size_t)4 * BATCH * 32;
  k_gru<<<dim3(BATCH), 256, 0, stream>>>(ru, ni, nh, deter_m,
                                         ln_r_g, ln_r_b, ln_u_g, ln_u_b, ln_n_g, ln_n_b,
                                         h_out, h_b);

  // --- prior head ---
  k_gemm_bf16<1><<<dim3(HID / BN, BATCH / BM), 256, SMEM_BYTES, stream>>>(
      h_b, DETER, nullptr, 0, wp1_b, bp1, p1_b, BATCH, HID);
  k_gemm_bf16<0><<<dim3(1, BATCH / BM), 256, SMEM_BYTES, stream>>>(
      p1_b, HID, nullptr, 0, wp2_b, bp2, prior, BATCH, 64);

  // --- posterior head (split-K A = [h | embedding]) ---
  k_gemm_bf16<1><<<dim3(HID / BN, BATCH / BM), 256, SMEM_BYTES, stream>>>(
      h_b, DETER, emb_b, EMBED, wq1_b, bq1, q1_b, BATCH, HID);
  k_gemm_bf16<0><<<dim3(1, BATCH / BM), 256, SMEM_BYTES, stream>>>(
      q1_b, HID, nullptr, 0, wq2_b, bq2, post, BATCH, 64);

  // --- distribution outputs ---
  k_heads<<<g1(B * 32), 256, 0, stream>>>(post, prior, out);

  (void)in_sizes; (void)n_in; (void)out_size; (void)ws_size;
}
